// AttentionEdge_58720792871446
// MI455X (gfx1250) — matrix operand order repaired and checked by reading, not compiled, not run on hardware
//
#include <hip/hip_runtime.h>
#include <hip/hip_bf16.h>

// ---- problem constants (from reference) ----
#define BB      16
#define SS      4
#define HID     4096
#define HH      32
#define KVH     8
#define DD      128
#define MAXSEQ  4096
#define SP      4092          // start_pos
#define NTOK    (BB*SS)       // 64 tokens

typedef __attribute__((ext_vector_type(16))) _Float16 v16h;
typedef __attribute__((ext_vector_type(8)))  _Float16 v8h;
typedef __attribute__((ext_vector_type(8)))  float    v8f;

union HF16 { v16h v; v8h h[2]; _Float16 e[16]; };

// -------------------------------------------------------------------------
// 1) fp32 -> f16 conversion of hidden_states
// -------------------------------------------------------------------------
__global__ void cvt_f32_f16(const float* __restrict__ in, _Float16* __restrict__ out, int n) {
    int i = blockIdx.x * blockDim.x + threadIdx.x;
    if (i < n) out[i] = (_Float16)in[i];
}

// -------------------------------------------------------------------------
// 2) Wave-tiled WMMA GEMM:  C[m,n] = sum_k A[m,k] * W[n,k] (+ bias[n])
//    A: f16 (M=64 x K), W: f32 row-major (N x K) converted to f16 on the fly.
//    One wave owns one 16x16 output tile; K-loop in steps of 32.
// -------------------------------------------------------------------------
__global__ __launch_bounds__(256)
void gemm_wmma(const _Float16* __restrict__ A, const float* __restrict__ W,
               const float* __restrict__ bias, float* __restrict__ C,
               int N, int K) {
    const int M = NTOK;
    const int wave = threadIdx.x >> 5;
    const int lane = threadIdx.x & 31;
    const int li   = lane & 15;
    const int hi   = lane >> 4;

    const int tilesM = M / 16;
    const int tile   = blockIdx.x * (blockDim.x >> 5) + wave;
    if (tile >= tilesM * (N / 16)) return;
    const int tm = tile % tilesM;
    const int tn = tile / tilesM;

    const int row = tm * 16 + li;
    const int col = tn * 16 + li;
    // A-fragment lane layout (16-bit A 16x32): lanes 0-15 K = 0..7 & 16..23,
    // lanes 16-31 K = 8..15 & 24..31  -> base offset hi*8.
    const _Float16* arow = A + (size_t)row * K + hi * 8;
    // B-fragment: lane holds W row `col`, K = hi*8+0..7 & 16+hi*8+0..7.
    const float* wrow = W + (size_t)col * K + hi * 8;

    v8f c = {};
    for (int k0 = 0; k0 < K; k0 += 32) {
        // stream the (once-read) weight rows ahead: global_prefetch_b8
        __builtin_prefetch(wrow + k0 + 256, 0, 1);

        HF16 a;
        a.h[0] = *(const v8h*)(arow + k0);
        a.h[1] = *(const v8h*)(arow + k0 + 16);

        HF16 b;
        const float4* f4 = (const float4*)(wrow + k0);
        float4 w0 = f4[0], w1 = f4[1], w2 = f4[4], w3 = f4[5];
        b.e[0]=(_Float16)w0.x;  b.e[1]=(_Float16)w0.y;  b.e[2]=(_Float16)w0.z;  b.e[3]=(_Float16)w0.w;
        b.e[4]=(_Float16)w1.x;  b.e[5]=(_Float16)w1.y;  b.e[6]=(_Float16)w1.z;  b.e[7]=(_Float16)w1.w;
        b.e[8]=(_Float16)w2.x;  b.e[9]=(_Float16)w2.y;  b.e[10]=(_Float16)w2.z; b.e[11]=(_Float16)w2.w;
        b.e[12]=(_Float16)w3.x; b.e[13]=(_Float16)w3.y; b.e[14]=(_Float16)w3.z; b.e[15]=(_Float16)w3.w;

        c = __builtin_amdgcn_wmma_f32_16x16x32_f16(false, a.v, false, b.v,
                                                   (short)0, c, false, false);
    }

    const float bv = bias ? bias[col] : 0.0f;
    #pragma unroll
    for (int r = 0; r < 8; ++r) {
        int orow = tm * 16 + r + hi * 8;      // C layout: row = r + 8*hi
        C[(size_t)orow * N + col] = c[r] + bv;
    }
}

// -------------------------------------------------------------------------
// 3) RoPE + repack into attention layouts.
//    qatt : f16 [B][KV][16][128]   (row = rep*4 + s,  h = kv*4 + rep)
//    katt/vatt : f16 [B][KV][4][128]
//    knew/vnew : f32 [B][KV][S][D]  (outputs 2 & 3)
// -------------------------------------------------------------------------
__global__ void rope_pack(const float* __restrict__ qraw, const float* __restrict__ kraw,
                          const float* __restrict__ vraw, const int* __restrict__ positions,
                          _Float16* __restrict__ qatt, _Float16* __restrict__ katt,
                          _Float16* __restrict__ vatt,
                          float* __restrict__ knew, float* __restrict__ vnew) {
    const int t = blockIdx.x;                 // token 0..63
    const int b = t >> 2, s = t & 3;
    const float pos = (float)positions[t];
    const float LOGB_OVER_HALF = 9.210340371976184f / 64.0f;   // ln(10000)/64

    // Q: H heads * 64 rotation pairs
    for (int idx = threadIdx.x; idx < HH * 64; idx += blockDim.x) {
        int h = idx >> 6, i = idx & 63;
        float ang = pos * __expf(-(float)i * LOGB_OVER_HALF);
        float cs = __cosf(ang), sn = __sinf(ang);
        const float* qb = qraw + (size_t)t * (HH * DD) + h * DD;
        float x1 = qb[i], x2 = qb[i + 64];
        float o1 = x1 * cs - x2 * sn;
        float o2 = x2 * cs + x1 * sn;
        int kv = h >> 2, rep = h & 3;
        int row = rep * 4 + s;
        _Float16* qd = qatt + (((size_t)(b * KVH + kv) * 16) + row) * DD;
        qd[i]      = (_Float16)o1;
        qd[i + 64] = (_Float16)o2;
    }
    // K: KV heads * 64 pairs (roped; duplicated to fp32 output + f16 ws)
    for (int idx = threadIdx.x; idx < KVH * 64; idx += blockDim.x) {
        int kv = idx >> 6, i = idx & 63;
        float ang = pos * __expf(-(float)i * LOGB_OVER_HALF);
        float cs = __cosf(ang), sn = __sinf(ang);
        const float* kb = kraw + (size_t)t * (KVH * DD) + kv * DD;
        float x1 = kb[i], x2 = kb[i + 64];
        float o1 = x1 * cs - x2 * sn;
        float o2 = x2 * cs + x1 * sn;
        size_t o = ((size_t)(b * KVH + kv) * SS + s) * DD;
        knew[o + i] = o1;      knew[o + i + 64] = o2;
        katt[o + i] = (_Float16)o1;  katt[o + i + 64] = (_Float16)o2;
    }
    // V: plain copy (fp32 output + f16 ws)
    for (int idx = threadIdx.x; idx < KVH * DD; idx += blockDim.x) {
        int kv = idx >> 7, i = idx & 127;
        float v = vraw[(size_t)t * (KVH * DD) + kv * DD + i];
        size_t o = ((size_t)(b * KVH + kv) * SS + s) * DD + i;
        vnew[o] = v;
        vatt[o] = (_Float16)v;
    }
}

// -------------------------------------------------------------------------
// 4) Flash-decode attention, key-split, async V staging.
//    grid = (B*KV, KSPLIT); AWAVES waves per block stream MAXSEQ/KSPLIT keys.
//    Per chunk: issue GLOBAL_LOAD_ASYNC_TO_LDS for the raw fp32 V tile, then
//    compute Q*K^T WMMAs + online softmax from K while the async engine fills
//    LDS; s_wait_asynccnt 0 before the P*V WMMAs read V from LDS.
// -------------------------------------------------------------------------
#define AWAVES 2
#define KSPLIT 8
#define VRAW_BYTES (AWAVES * 32 * 128 * 4)          // fp32 V tiles, offset 0 in LDS
#define PT_BYTES   (AWAVES * 16 * 32 * 2)
__global__ __launch_bounds__(AWAVES * 32)
void attn_partial(const _Float16* __restrict__ qatt, const _Float16* __restrict__ katt,
                  const _Float16* __restrict__ vatt,
                  const float* __restrict__ kcache, const float* __restrict__ vcache,
                  float* __restrict__ pO   /* [BKV][KSPLIT][16][128] */,
                  float* __restrict__ pML  /* [BKV][KSPLIT][2][16]   */) {
    // LDS: raw fp32 V tile per wave (32x128), P tile (16x32 f16), m/l stats.
    // Vraw region is reused as fp32 O partials for the wave merge.
    __shared__ __align__(16) unsigned char smem_raw[VRAW_BYTES + PT_BYTES + AWAVES*32*4];
    float*    Vraw_base = (float*)smem_raw;                    // LDS offset 0
    _Float16* Pt_base   = (_Float16*)(smem_raw + VRAW_BYTES);
    float*    ml_base   = (float*)(smem_raw + VRAW_BYTES + PT_BYTES);

    const int bkv   = blockIdx.x;
    const int split = blockIdx.y;
    const int wave  = threadIdx.x >> 5;
    const int lane  = threadIdx.x & 31;
    const int li = lane & 15, hi = lane >> 4;

    float*    Vr = Vraw_base + wave * 32 * 128;
    _Float16* Pt = Pt_base + wave * 16 * 32;
    const unsigned vlds_base = (unsigned)(wave * 32 * 128 * 4) + (unsigned)lane * 512u;

    // Q fragments (4 x K=32 d-chunks), resident for whole key loop.
    HF16 qa[4];
    {
        const _Float16* qr = qatt + (size_t)bkv * 16 * DD + li * DD + hi * 16;
        #pragma unroll
        for (int dc = 0; dc < 4; ++dc) {
            qa[dc].h[0] = *(const v8h*)(qr + dc * 32);
            qa[dc].h[1] = *(const v8h*)(qr + dc * 32 + 8);
        }
    }
    const float*    kc = kcache + (size_t)bkv * MAXSEQ * DD;
    const float*    vc = vcache + (size_t)bkv * MAXSEQ * DD;
    const _Float16* kn = katt + (size_t)bkv * SS * DD;
    const _Float16* vn = vatt + (size_t)bkv * SS * DD;

    float m_run[8], l_run[8];
    v8f acc[8] = {};                    // 8 d-tiles, each v8f over rows
    #pragma unroll
    for (int r = 0; r < 8; ++r) { m_run[r] = -1e30f; l_run[r] = 0.0f; }

    const float scale = 0.08838834764831845f;   // 1/sqrt(128)

    const int cbeg = split * (MAXSEQ / 32 / KSPLIT);
    const int cend = cbeg + (MAXSEQ / 32 / KSPLIT);

    for (int ci = cbeg + wave; ci < cend; ci += AWAVES) {
        const int t0 = ci * 32;

        // previous iteration's LDS reads must retire before the async engine
        // overwrites this wave's Vraw tile
        asm volatile("s_wait_dscnt 0x0" ::: "memory");

        // ---- async-stage raw fp32 V tile: one key row per lane ------------
        {
            int t = t0 + lane;
            if (t < SP) {
                unsigned long long src = (unsigned long long)(const void*)(vc + (size_t)t * DD);
                #pragma unroll
                for (int it = 0; it < 32; ++it) {
                    unsigned dst = vlds_base + (unsigned)(it * 16);
                    unsigned long long s64 = src + (unsigned)(it * 16);
                    asm volatile("global_load_async_to_lds_b128 %0, %1, off"
                                 :: "v"(dst), "v"(s64) : "memory");
                }
            } else {
                // new keys (<= 4 lanes, last chunk only): plain LDS stores
                const _Float16* vp = vn + (size_t)(t - SP) * DD;
                float* dr = Vr + lane * 128;
                for (int d = 0; d < DD; ++d) dr[d] = (float)vp[d];
            }
        }

        // ---- prefetch this wave's NEXT chunk of K (global_prefetch_b8) ----
        if (ci + AWAVES < cend) {
            const int tp = (ci + AWAVES) * 32 + lane;
            __builtin_prefetch(kc + (size_t)tp * DD, 0, 1);
            __builtin_prefetch(kc + (size_t)tp * DD + 64, 0, 1);
        }

        // ---- scores: two 16-key tiles, K over d (4 x 32) ------------------
        v8f sc[2];
        #pragma unroll
        for (int tt = 0; tt < 2; ++tt) {
            int t = t0 + tt * 16 + li;
            HF16 kb[4];
            if (t < SP) {
                const float* kp = kc + (size_t)t * DD + hi * 16;
                #pragma unroll
                for (int dc = 0; dc < 4; ++dc) {
                    const float4* f4 = (const float4*)(kp + dc * 32);
                    float4 w0 = f4[0], w1 = f4[1], w2 = f4[2], w3 = f4[3];
                    kb[dc].e[0]=(_Float16)w0.x;  kb[dc].e[1]=(_Float16)w0.y;
                    kb[dc].e[2]=(_Float16)w0.z;  kb[dc].e[3]=(_Float16)w0.w;
                    kb[dc].e[4]=(_Float16)w1.x;  kb[dc].e[5]=(_Float16)w1.y;
                    kb[dc].e[6]=(_Float16)w1.z;  kb[dc].e[7]=(_Float16)w1.w;
                    kb[dc].e[8]=(_Float16)w2.x;  kb[dc].e[9]=(_Float16)w2.y;
                    kb[dc].e[10]=(_Float16)w2.z; kb[dc].e[11]=(_Float16)w2.w;
                    kb[dc].e[12]=(_Float16)w3.x; kb[dc].e[13]=(_Float16)w3.y;
                    kb[dc].e[14]=(_Float16)w3.z; kb[dc].e[15]=(_Float16)w3.w;
                }
            } else {
                const _Float16* kp = kn + (size_t)(t - SP) * DD + hi * 16;
                #pragma unroll
                for (int dc = 0; dc < 4; ++dc) {
                    kb[dc].h[0] = *(const v8h*)(kp + dc * 32);
                    kb[dc].h[1] = *(const v8h*)(kp + dc * 32 + 8);
                }
            }
            // WMMA outside divergent flow (EXEC must be all-ones)
            v8f c = {};
            #pragma unroll
            for (int dc = 0; dc < 4; ++dc)
                c = __builtin_amdgcn_wmma_f32_16x16x32_f16(false, qa[dc].v, false, kb[dc].v,
                                                           (short)0, c, false, false);
            sc[tt] = c;
        }

        // ---- online softmax update (row stats across 16-lane groups) ------
        #pragma unroll
        for (int tt = 0; tt < 2; ++tt)
            #pragma unroll
            for (int r = 0; r < 8; ++r) sc[tt][r] *= scale;

        float mnew[8], fac[8];
        #pragma unroll
        for (int r = 0; r < 8; ++r) {
            float v = fmaxf(sc[0][r], sc[1][r]);
            v = fmaxf(v, __shfl_xor(v, 1));
            v = fmaxf(v, __shfl_xor(v, 2));
            v = fmaxf(v, __shfl_xor(v, 4));
            v = fmaxf(v, __shfl_xor(v, 8));
            mnew[r] = fmaxf(m_run[r], v);
            fac[r]  = __expf(m_run[r] - mnew[r]);
            m_run[r] = mnew[r];
        }
        #pragma unroll
        for (int r = 0; r < 8; ++r) {
            float p0 = __expf(sc[0][r] - mnew[r]);
            float p1 = __expf(sc[1][r] - mnew[r]);
            sc[0][r] = p0; sc[1][r] = p1;
            float s = p0 + p1;
            s += __shfl_xor(s, 1);
            s += __shfl_xor(s, 2);
            s += __shfl_xor(s, 4);
            s += __shfl_xor(s, 8);
            l_run[r] = l_run[r] * fac[r] + s;
        }
        #pragma unroll
        for (int dt = 0; dt < 8; ++dt)
            #pragma unroll
            for (int r = 0; r < 8; ++r) acc[dt][r] *= fac[r];

        // ---- P (C-layout) -> LDS -> A-layout fragment ---------------------
        #pragma unroll
        for (int r = 0; r < 8; ++r) {
            int prow = r + hi * 8;
            Pt[prow * 32 + li]      = (_Float16)sc[0][r];
            Pt[prow * 32 + 16 + li] = (_Float16)sc[1][r];
        }
        asm volatile("s_wait_dscnt 0x0" ::: "memory");
        HF16 pa;
        {
            const _Float16* pr = Pt + li * 32 + hi * 16;
            pa.h[0] = *(const v8h*)(pr);
            pa.h[1] = *(const v8h*)(pr + 8);
        }

        // ---- V tile must be resident in LDS now ---------------------------
        asm volatile("s_wait_asynccnt 0x0" ::: "memory");

        // ---- P·V : 8 d-tiles, B gathered from raw fp32 V in LDS -----------
        #pragma unroll
        for (int dt = 0; dt < 8; ++dt) {
            HF16 vb;
            const float* vcol = Vr + hi * 16 * 128 + dt * 16 + li;
            #pragma unroll
            for (int j = 0; j < 16; ++j)
                vb.e[j] = (_Float16)vcol[j * 128];
            acc[dt] = __builtin_amdgcn_wmma_f32_16x16x32_f16(false, pa.v, false, vb.v,
                                                             (short)0, acc[dt], false, false);
        }
    }

    // ---- merge the AWAVES partials via LDS, emit block partial ------------
    asm volatile("s_wait_dscnt 0x0" ::: "memory");
    float* sm_m = ml_base + wave * 32;       // [16] m then [16] l
    float* sm_l = sm_m + 16;
    float* sm_O = (float*)Vraw_base;         // reuse: [AWAVES][16][128] f32
    #pragma unroll
    for (int r = 0; r < 8; ++r) {
        int row = r + hi * 8;
        if (li == 0) { sm_m[row] = m_run[r]; sm_l[row] = l_run[r]; }
        #pragma unroll
        for (int dt = 0; dt < 8; ++dt)
            sm_O[((size_t)wave * 16 + row) * 128 + dt * 16 + li] = acc[dt][r];
    }
    __syncthreads();

    const size_t pbase = (size_t)bkv * KSPLIT + split;
    for (int e = threadIdx.x; e < 16 * 128; e += blockDim.x) {
        int row = e >> 7, d = e & 127;
        float M = -1e30f;
        for (int w = 0; w < AWAVES; ++w) M = fmaxf(M, ml_base[w * 32 + row]);
        float L = 0.0f, o = 0.0f;
        for (int w = 0; w < AWAVES; ++w) {
            float f = __expf(ml_base[w * 32 + row] - M);
            L += ml_base[w * 32 + 16 + row] * f;
            o += sm_O[((size_t)w * 16 + row) * 128 + d] * f;
        }
        pO[(pbase * 16 + row) * 128 + d] = o;          // unnormalized
        if (d == 0) {
            pML[pbase * 32 + row]      = M;
            pML[pbase * 32 + 16 + row] = L;
        }
    }
}

// -------------------------------------------------------------------------
// 5) merge the KSPLIT partials per (b,kv) -> f16 attention output
// -------------------------------------------------------------------------
__global__ __launch_bounds__(128)
void attn_merge(const float* __restrict__ pO, const float* __restrict__ pML,
                _Float16* __restrict__ attn_out /* f16 [tok][H][128] */) {
    const int bkv = blockIdx.x;
    const int b = bkv / KVH, kv = bkv % KVH;
    for (int e = threadIdx.x; e < 16 * 128; e += blockDim.x) {
        int row = e >> 7, d = e & 127;
        float M = -1e30f;
        #pragma unroll
        for (int sp = 0; sp < KSPLIT; ++sp)
            M = fmaxf(M, pML[((size_t)bkv * KSPLIT + sp) * 32 + row]);
        float L = 0.0f, o = 0.0f;
        #pragma unroll
        for (int sp = 0; sp < KSPLIT; ++sp) {
            size_t pb = (size_t)bkv * KSPLIT + sp;
            float f = __expf(pML[pb * 32 + row] - M);
            L += pML[pb * 32 + 16 + row] * f;
            o += pO[(pb * 16 + row) * 128 + d] * f;
        }
        int rep = row >> 2, s = row & 3;
        int h = kv * 4 + rep;
        int tok = b * SS + s;
        attn_out[((size_t)tok * HH + h) * DD + d] = (_Float16)(o / L);
    }
}

// -------------------------------------------------------------------------
// launcher
// -------------------------------------------------------------------------
extern "C" void kernel_launch(void* const* d_in, const int* in_sizes, int n_in,
                              void* d_out, int out_size, void* d_ws, size_t ws_size,
                              hipStream_t stream) {
    const float* hs    = (const float*)d_in[0];
    const int*   pos   = (const int*)  d_in[1];
    const float* kcach = (const float*)d_in[2];
    const float* vcach = (const float*)d_in[3];
    const float* q_w   = (const float*)d_in[5];
    const float* q_b   = (const float*)d_in[6];
    const float* k_w   = (const float*)d_in[7];
    const float* k_b   = (const float*)d_in[8];
    const float* v_w   = (const float*)d_in[9];
    const float* v_b   = (const float*)d_in[10];
    const float* o_w   = (const float*)d_in[11];

    float* out  = (float*)d_out;                // [64][4096]
    float* knew = out + (size_t)NTOK * HID;     // [B][KV][S][D]
    float* vnew = knew + (size_t)BB * KVH * SS * DD;

    // workspace carve-out
    char* ws = (char*)d_ws;
    _Float16* hs16   = (_Float16*)ws; ws += (size_t)NTOK * HID * 2;
    float*    qraw   = (float*)ws;    ws += (size_t)NTOK * HH  * DD * 4;
    float*    kraw   = (float*)ws;    ws += (size_t)NTOK * KVH * DD * 4;
    float*    vraw   = (float*)ws;    ws += (size_t)NTOK * KVH * DD * 4;
    _Float16* qatt   = (_Float16*)ws; ws += (size_t)BB * KVH * 16 * DD * 2;
    _Float16* katt   = (_Float16*)ws; ws += (size_t)BB * KVH * SS * DD * 2;
    _Float16* vatt   = (_Float16*)ws; ws += (size_t)BB * KVH * SS * DD * 2;
    _Float16* attn16 = (_Float16*)ws; ws += (size_t)NTOK * HID * 2;
    float*    pO     = (float*)ws;    ws += (size_t)BB * KVH * KSPLIT * 16 * DD * 4;
    float*    pML    = (float*)ws;    ws += (size_t)BB * KVH * KSPLIT * 32 * 4;

    // 1) hidden -> f16
    {
        int n = NTOK * HID;
        cvt_f32_f16<<<(n + 255) / 256, 256, 0, stream>>>(hs, hs16, n);
    }
    // 2) projections: q (N=4096), k (N=1024), v (N=1024)
    gemm_wmma<<<(4 * (HID / 16)) / 8, 256, 0, stream>>>(hs16, q_w, q_b, qraw, HH * DD, HID);
    gemm_wmma<<<(4 * ((KVH * DD) / 16)) / 8, 256, 0, stream>>>(hs16, k_w, k_b, kraw, KVH * DD, HID);
    gemm_wmma<<<(4 * ((KVH * DD) / 16)) / 8, 256, 0, stream>>>(hs16, v_w, v_b, vraw, KVH * DD, HID);
    // 3) rope + pack (also writes k_new / v_new fp32 outputs)
    rope_pack<<<NTOK, 256, 0, stream>>>(qraw, kraw, vraw, pos, qatt, katt, vatt, knew, vnew);
    // 4) attention: key-split partials (async V staging), then merge
    attn_partial<<<dim3(BB * KVH, KSPLIT), AWAVES * 32, 0, stream>>>(
        qatt, katt, vatt, kcach, vcach, pO, pML);
    attn_merge<<<BB * KVH, 128, 0, stream>>>(pO, pML, attn16);
    // 5) o-projection -> out
    gemm_wmma<<<(4 * (HID / 16)) / 8, 256, 0, stream>>>(attn16, o_w, nullptr, out, HID, HID);
}
